// BertCLModel_48627619725603
// MI455X (gfx1250) — matrix operand order repaired and checked by reading, not compiled, not run on hardware
//
#include <hip/hip_runtime.h>
#include <hip/hip_bf16.h>

typedef __attribute__((ext_vector_type(16))) _Float16 v16h;
typedef __attribute__((ext_vector_type(8)))  _Float16 v8h;
typedef __attribute__((ext_vector_type(8)))  float    v8f;

#define B_ROWS 512
#define D_DIM  768
#define H_DIM  256
#define NPAIRS 57280   // sum_{i=0}^{127} (511 - i)

// ---------------------------------------------------------------------------
// WMMA fragment loaders (CDNA5 16x16x32 f16 layouts, ISA 7.12.2)
// A (16x32, MxK): lane<16 -> M=lane, K = k0..k0+7 (v0-3) and k0+16..k0+23 (v4-7), k0=kbase
//                 lane>=16 -> M=lane-16, same with k0=kbase+8
// B (32x16, KxN): same K mapping as A; lane<16 -> N=lane, k0=kbase ; lane>=16 -> N=lane-16, k0=kbase+8
// C/D (16x16):    vgpr r, lane<16 -> M=r, N=lane ; lane>=16 -> M=8+r, N=lane-16
// ---------------------------------------------------------------------------
static __device__ __forceinline__ v16h load_a_frag(const _Float16* base, int ld, int kbase) {
    int lane = threadIdx.x & 31;
    const _Float16* p = base + (lane & 15) * ld + kbase + ((lane & 16) ? 8 : 0);
    v8h lo = *(const v8h*)p;          // K = k0 .. k0+7
    v8h hi = *(const v8h*)(p + 16);   // K = k0+16 .. k0+23
    v16h r;
#pragma unroll
    for (int t = 0; t < 8; ++t) { r[t] = lo[t]; r[t + 8] = hi[t]; }
    return r;
}

static __device__ __forceinline__ v16h load_b_frag(const _Float16* base, int ld, int kbase) {
    int lane = threadIdx.x & 31;
    const _Float16* p = base + (lane & 15) * ld + kbase + ((lane & 16) ? 8 : 0);
    v8h lo = *(const v8h*)p;          // K = k0 .. k0+7
    v8h hi = *(const v8h*)(p + 16);   // K = k0+16 .. k0+23
    v16h r;
#pragma unroll
    for (int t = 0; t < 8; ++t) { r[t] = lo[t]; r[t + 8] = hi[t]; }
    return r;
}

static __device__ __forceinline__ void store_c_frag(float* C, int ldc, int mbase, int nbase, v8f c) {
    int lane = threadIdx.x & 31;
    int col  = nbase + (lane & 15);
    int row0 = mbase + ((lane & 16) ? 8 : 0);
#pragma unroll
    for (int r = 0; r < 8; ++r) C[(row0 + r) * ldc + col] = c[r];
}

// ---------------------------------------------------------------------------
// 0) zero accumulators
__global__ void zero_kernel(float* accum) {
    if (threadIdx.x == 0) { accum[0] = 0.f; accum[1] = 0.f; }
}

// 1) row L2-normalize -> f16   (one block per row, 256 threads, 768 = 3*256)
__global__ __launch_bounds__(256) void normalize_kernel(const float* __restrict__ emb,
                                                        _Float16* __restrict__ zf16) {
    __shared__ float red[256];
    int row = blockIdx.x, tid = threadIdx.x;
    const float* p = emb + row * D_DIM;
    float x0 = p[tid], x1 = p[tid + 256], x2 = p[tid + 512];
    red[tid] = x0 * x0 + x1 * x1 + x2 * x2;
    __syncthreads();
    for (int w = 128; w; w >>= 1) { if (tid < w) red[tid] += red[tid + w]; __syncthreads(); }
    float scale = 1.f / fmaxf(sqrtf(red[0]), 1e-12f);
    _Float16* o = zf16 + row * D_DIM;
    o[tid]       = (_Float16)(x0 * scale);
    o[tid + 256] = (_Float16)(x1 * scale);
    o[tid + 512] = (_Float16)(x2 * scale);
}

// 2) W1, W2 -> f16
__global__ void convert_kernel(const float* __restrict__ W1, const float* __restrict__ W2,
                               _Float16* __restrict__ w1h, _Float16* __restrict__ w2h) {
    int idx = blockIdx.x * 256 + threadIdx.x;
    const int N1 = H_DIM * 2 * D_DIM;          // 393216
    const int N2 = H_DIM * H_DIM;              // 65536
    if (idx < N1)            w1h[idx]      = (_Float16)W1[idx];
    else if (idx < N1 + N2)  w2h[idx - N1] = (_Float16)W2[idx - N1];
}

// 3) S = z @ z^T   [512x512] f32 ; one wave per 16x64 tile (A-frag reused 4x),
//    32 M-tiles x 8 N-supertiles = 256 waves = 32 blocks x 8 waves
__global__ __launch_bounds__(256) void gemm_s_kernel(const _Float16* __restrict__ z,
                                                     float* __restrict__ S) {
    int wave = (blockIdx.x * blockDim.x + threadIdx.x) >> 5;   // 0..255
    int tm  = (wave >> 3) << 4;    // 0..496 step 16
    int tn4 = (wave & 7) << 6;     // 0..448 step 64
    v8f zero = {};
    v8f acc[4] = {zero, zero, zero, zero};
    for (int kb = 0; kb < D_DIM; kb += 32) {
        // issue all fragment loads first, then the WMMA chain (overlap loads w/ matrix pipe)
        v16h a  = load_a_frag(z + tm * D_DIM, D_DIM, kb);
        v16h b0 = load_b_frag(z + (tn4 +  0) * D_DIM, D_DIM, kb);
        v16h b1 = load_b_frag(z + (tn4 + 16) * D_DIM, D_DIM, kb);
        v16h b2 = load_b_frag(z + (tn4 + 32) * D_DIM, D_DIM, kb);
        v16h b3 = load_b_frag(z + (tn4 + 48) * D_DIM, D_DIM, kb);
        acc[0] = __builtin_amdgcn_wmma_f32_16x16x32_f16(false, a, false, b0, (short)0, acc[0], false, false);
        acc[1] = __builtin_amdgcn_wmma_f32_16x16x32_f16(false, a, false, b1, (short)0, acc[1], false, false);
        acc[2] = __builtin_amdgcn_wmma_f32_16x16x32_f16(false, a, false, b2, (short)0, acc[2], false, false);
        acc[3] = __builtin_amdgcn_wmma_f32_16x16x32_f16(false, a, false, b3, (short)0, acc[3], false, false);
    }
#pragma unroll
    for (int u = 0; u < 4; ++u) store_c_frag(S, B_ROWS, tm, tn4 + u * 16, acc[u]);
}

// 4) log_denom[i] = log( sum_{k != i} exp(2*S[i,k]) )
__global__ __launch_bounds__(256) void logdenom_kernel(const float* __restrict__ S,
                                                       float* __restrict__ logden) {
    __shared__ float red[256];
    int row = blockIdx.x, tid = threadIdx.x;
    float s = 0.f;
    for (int k = tid; k < B_ROWS; k += 256)
        if (k != row) s += expf(2.f * S[row * B_ROWS + k]);
    red[tid] = s;
    __syncthreads();
    for (int w = 128; w; w >>= 1) { if (tid < w) red[tid] += red[tid + w]; __syncthreads(); }
    if (tid == 0) logden[row] = logf(red[0]);
}

// 5) closs_sum = sum_{i<j<128} (logden[i] - 2*S[i,j])  -> accum[0]
__global__ __launch_bounds__(128) void closs_kernel(const float* __restrict__ S,
                                                    const float* __restrict__ logden,
                                                    float* __restrict__ accum) {
    __shared__ float red[128];
    int i = blockIdx.x, j = threadIdx.x;
    float v = 0.f;
    if (j > i) v = logden[i] - 2.f * S[i * B_ROWS + j];
    red[j] = v;
    __syncthreads();
    for (int w = 64; w; w >>= 1) { if (j < w) red[j] += red[j + w]; __syncthreads(); }
    if (j == 0) atomicAdd(&accum[0], red[0]);
}

// 6) AB[r][h] = dot(z[r], W1[h, 0:768]) for h<256 ; AB[r][256+h] = dot(z[r], W1[h, 768:1536])
//    Same 16x64-per-wave blocking; koff uniform per 64-wide supertile (256 % 64 == 0).
__global__ __launch_bounds__(256) void gemm_ab_kernel(const _Float16* __restrict__ z,
                                                      const _Float16* __restrict__ w1h,
                                                      float* __restrict__ AB) {
    int wave = (blockIdx.x * blockDim.x + threadIdx.x) >> 5;
    int tm  = (wave >> 3) << 4;
    int tn4 = (wave & 7) << 6;
    int koff = (tn4 >= H_DIM) ? D_DIM : 0;
    const _Float16* bb = w1h + (tn4 & (H_DIM - 1)) * (2 * D_DIM) + koff;
    v8f zero = {};
    v8f acc[4] = {zero, zero, zero, zero};
    for (int kb = 0; kb < D_DIM; kb += 32) {
        v16h a  = load_a_frag(z + tm * D_DIM, D_DIM, kb);
        v16h b0 = load_b_frag(bb +  0 * (2 * D_DIM), 2 * D_DIM, kb);
        v16h b1 = load_b_frag(bb + 16 * (2 * D_DIM), 2 * D_DIM, kb);
        v16h b2 = load_b_frag(bb + 32 * (2 * D_DIM), 2 * D_DIM, kb);
        v16h b3 = load_b_frag(bb + 48 * (2 * D_DIM), 2 * D_DIM, kb);
        acc[0] = __builtin_amdgcn_wmma_f32_16x16x32_f16(false, a, false, b0, (short)0, acc[0], false, false);
        acc[1] = __builtin_amdgcn_wmma_f32_16x16x32_f16(false, a, false, b1, (short)0, acc[1], false, false);
        acc[2] = __builtin_amdgcn_wmma_f32_16x16x32_f16(false, a, false, b2, (short)0, acc[2], false, false);
        acc[3] = __builtin_amdgcn_wmma_f32_16x16x32_f16(false, a, false, b3, (short)0, acc[3], false, false);
    }
#pragma unroll
    for (int u = 0; u < 4; ++u) store_c_frag(AB, 2 * H_DIM, tm, tn4 + u * 16, acc[u]);
}

// 7) fused pair MLP: h1 = relu(AB[i,:256]+AB[j,256:]+b1) -> WMMA vs W2 -> relu(+b2)
//    -> dot W3 (+b3) -> BCE sum into accum[1].  4 waves/block, 16 pairs/wave,
//    895 blocks * 64 pairs = 57280 = NPAIRS exactly.
__global__ __launch_bounds__(128) void pair_kernel(const float* __restrict__ AB,
                                                   const float* __restrict__ b1,
                                                   const _Float16* __restrict__ w2h,
                                                   const float* __restrict__ b2,
                                                   const float* __restrict__ W3,
                                                   const float* __restrict__ b3,
                                                   float* __restrict__ accum) {
    __shared__ __align__(16) _Float16 h1[4][16][264];   // 264 = 256 + 8 pad (bank rotate)
    __shared__ float bsum;
    int tid = threadIdx.x, wave = tid >> 5, lane = tid & 31;
    if (tid == 0) bsum = 0.f;

    int pbase = blockIdx.x * 64 + wave * 16;

    // every lane inverts the triangular index for pair m = lane & 15
    // cum(i) = i*(1023 - i)/2  (pairs consumed before row i)
    int p = pbase + (lane & 15);
    int iv = 0, jv = 0;
    if (p < NPAIRS) {
        float pf = (float)p;
        int i = (int)((1023.0f - sqrtf(1023.0f * 1023.0f - 8.0f * pf)) * 0.5f);
        if (i < 0) i = 0;
        if (i > 127) i = 127;
        while (i > 0   && i * (1023 - i) / 2 > p) --i;
        while (i < 127 && (i + 1) * (1022 - i) / 2 <= p) ++i;
        iv = i;
        jv = i + 1 + (p - i * (1023 - i) / 2);
    }

    // warm L0 with the first W2 slice (global_prefetch_b8; 8 calls x 32 lanes x 128B = 32KB)
#pragma unroll
    for (int t = 0; t < 8; ++t)
        __builtin_prefetch((const char*)w2h + (lane + 32 * t) * 128, 0, 3);

    // build h1 tile for this wave's 16 pairs
    float b1r[8];
#pragma unroll
    for (int t = 0; t < 8; ++t) b1r[t] = b1[lane + 32 * t];
    for (int m = 0; m < 16; ++m) {
        int im = __shfl(iv, m, 32);
        int jm = __shfl(jv, m, 32);
        const float* pi = AB + im * (2 * H_DIM);
        const float* pj = AB + jm * (2 * H_DIM) + H_DIM;
#pragma unroll
        for (int t = 0; t < 8; ++t) {
            int k = lane + 32 * t;
            float v = fmaxf(pi[k] + pj[k] + b1r[t], 0.f);
            h1[wave][m][k] = (_Float16)v;
        }
    }
    __syncthreads();

    // layer 2 + 3: 4 N-supertiles, A-frag reused across 4 WMMAs, epilogue fused
    const _Float16* h1base = &h1[wave][0][0];
    float logacc[8];
#pragma unroll
    for (int r = 0; r < 8; ++r) logacc[r] = 0.f;

    for (int nt4 = 0; nt4 < H_DIM; nt4 += 64) {
        // prefetch next 32KB W2 slice while this one computes
        if (nt4 + 64 < H_DIM) {
#pragma unroll
            for (int t = 0; t < 8; ++t)
                __builtin_prefetch((const char*)(w2h + (nt4 + 64) * H_DIM) + (lane + 32 * t) * 128, 0, 3);
        }
        const _Float16* wb = w2h + nt4 * H_DIM;
        v8f zero = {};
        v8f acc[4] = {zero, zero, zero, zero};
        for (int kb = 0; kb < H_DIM; kb += 32) {
            // A from LDS, 4 B-frags from global: issue loads together, then WMMA chain
            v16h a  = load_a_frag(h1base, 264, kb);
            v16h b0 = load_b_frag(wb +  0 * H_DIM, H_DIM, kb);
            v16h b1f = load_b_frag(wb + 16 * H_DIM, H_DIM, kb);
            v16h b2f = load_b_frag(wb + 32 * H_DIM, H_DIM, kb);
            v16h b3f = load_b_frag(wb + 48 * H_DIM, H_DIM, kb);
            acc[0] = __builtin_amdgcn_wmma_f32_16x16x32_f16(false, a, false, b0,  (short)0, acc[0], false, false);
            acc[1] = __builtin_amdgcn_wmma_f32_16x16x32_f16(false, a, false, b1f, (short)0, acc[1], false, false);
            acc[2] = __builtin_amdgcn_wmma_f32_16x16x32_f16(false, a, false, b2f, (short)0, acc[2], false, false);
            acc[3] = __builtin_amdgcn_wmma_f32_16x16x32_f16(false, a, false, b3f, (short)0, acc[3], false, false);
        }
#pragma unroll
        for (int u = 0; u < 4; ++u) {
            int col = nt4 + u * 16 + (lane & 15);
            float b2c = b2[col];
            float w3c = W3[col];
#pragma unroll
            for (int r = 0; r < 8; ++r) {
                float h2 = fmaxf(acc[u][r] + b2c, 0.f);
                logacc[r] += w3c * h2;
            }
        }
    }

    // reduce each reg across the 16-lane half-wave (lanes 0-15: pairs 0-7, 16-31: pairs 8-15)
#pragma unroll
    for (int r = 0; r < 8; ++r) {
        float v = logacc[r];
        for (int off = 8; off; off >>= 1) v += __shfl_xor(v, off, 16);
        logacc[r] = v;
    }

    // lanes 0-7 own pairs 0-7 (reg = lane); lanes 16-23 own pairs 8-15 (reg = lane-16)
    int rsel = lane & 7;
    int mb   = (lane & 16) ? (8 + (lane & 7)) : (lane & 7);
    int jm_b = __shfl(jv, mb, 32);              // pair mb held by lane mb
    bool active = ((lane & 8) == 0) && (pbase + mb < NPAIRS);
    float myl = 0.f;
#pragma unroll
    for (int r = 0; r < 8; ++r) myl = (r == rsel) ? logacc[r] : myl;
    if (active) {
        float l = myl + b3[0];
        float label = (jm_b < 256) ? 1.f : 0.f;
        float bce = fmaxf(l, 0.f) - l * label + log1pf(expf(-fabsf(l)));
        atomicAdd(&bsum, bce);
    }
    __syncthreads();
    if (tid == 0) atomicAdd(&accum[1], bsum);
}

// 8) out = -2*(n-1)/n * closs_sum + bce_sum / P
__global__ void finalize_kernel(const float* __restrict__ accum, float* __restrict__ out) {
    out[0] = (-2.f * 127.f / 128.f) * accum[0] + accum[1] / (float)NPAIRS;
}

// ---------------------------------------------------------------------------
extern "C" void kernel_launch(void* const* d_in, const int* in_sizes, int n_in,
                              void* d_out, int out_size, void* d_ws, size_t ws_size,
                              hipStream_t stream) {
    const float* emb = (const float*)d_in[0];
    const float* W1  = (const float*)d_in[1];
    const float* b1  = (const float*)d_in[2];
    const float* W2  = (const float*)d_in[3];
    const float* b2  = (const float*)d_in[4];
    const float* W3  = (const float*)d_in[5];
    const float* b3  = (const float*)d_in[6];
    float* out = (float*)d_out;

    char* w = (char*)d_ws;
    float*    accum  = (float*)(w + 0);            //      16 B (2 floats used)
    _Float16* zf16   = (_Float16*)(w + 16);        //  786432 B  [512*768] f16
    _Float16* w1h    = (_Float16*)(w + 786448);    //  786432 B  [256*1536] f16
    _Float16* w2h    = (_Float16*)(w + 1572880);   //  131072 B  [256*256] f16
    float*    S      = (float*)(w + 1703952);      // 1048576 B  [512*512] f32
    float*    logden = (float*)(w + 2752528);      //    2048 B  [512] f32
    float*    AB     = (float*)(w + 2754576);      // 1048576 B  [512*512] f32
    // total 3,803,152 B

    zero_kernel<<<1, 32, 0, stream>>>(accum);
    normalize_kernel<<<B_ROWS, 256, 0, stream>>>(emb, zf16);
    convert_kernel<<<1792, 256, 0, stream>>>(W1, W2, w1h, w2h);     // 458752 elems
    gemm_s_kernel<<<32, 256, 0, stream>>>(zf16, S);                 // 256 waves, 16x64 tiles
    logdenom_kernel<<<B_ROWS, 256, 0, stream>>>(S, logden);
    closs_kernel<<<128, 128, 0, stream>>>(S, logden, accum);
    gemm_ab_kernel<<<32, 256, 0, stream>>>(zf16, w1h, AB);
    pair_kernel<<<895, 128, 0, stream>>>(AB, b1, w2h, b2, W3, b3, accum);
    finalize_kernel<<<1, 1, 0, stream>>>(accum, out);
}
